// Block_21157008900211
// MI455X (gfx1250) — matrix operand order repaired and checked by reading, not compiled, not run on hardware
//
#include <hip/hip_runtime.h>
#include <hip/hip_bf16.h>
#include <math.h>

#define DIM    768
#define NH     12
#define HD     64
#define HIDDEN 3072
#define BB     16
#define TT     577
#define MROWS  (BB*TT)      // 9232
#define TPAD   608          // keys padded to 19 chunks of 32
#define QTILES 37           // ceil(577/16)
#define EPSF   1e-6f

typedef _Float16 v16h __attribute__((ext_vector_type(16)));
typedef _Float16 h8   __attribute__((ext_vector_type(8)));
typedef float    v8f  __attribute__((ext_vector_type(8)));

union F16x16 { v16h v; h8 h[2]; };

// A-matrix fragment (16x32 f16, M x K), row-major source, per ISA 7.12.2:
// lane l: M = l&15, half hl = l>>4; VGPR 0..3 <- K = 8*hl + (0..7), VGPR 4..7 <- K = 16+8*hl+(0..7)
__device__ __forceinline__ v16h ld_afrag(const _Float16* rowp, int hl) {
  F16x16 u;
  u.h[0] = *(const h8*)(rowp + 8 * hl);
  u.h[1] = *(const h8*)(rowp + 16 + 8 * hl);
  return u.v;
}
// B-matrix fragment (32x16 f16, K x N) loaded from B-transposed [N,K] row-major:
// lane l: N = l&15; VGPR 0..3 <- K = 8*hl + (0..7), VGPR 4..7 <- K = 16+8*hl+(0..7)
__device__ __forceinline__ v16h ld_btfrag(const _Float16* rowp, int hl) {
  F16x16 u;
  u.h[0] = *(const h8*)(rowp + 8 * hl);
  u.h[1] = *(const h8*)(rowp + 16 + 8 * hl);
  return u.v;
}

#define WMMA(a, b, c) __builtin_amdgcn_wmma_f32_16x16x32_f16(false, (a), false, (b), (short)0, (c), false, false)

// ---------------- weight transpose + fp32 -> f16 ----------------
__global__ __launch_bounds__(256) void convT_kernel(const float* __restrict__ W,
                                                    _Float16* __restrict__ Wt,
                                                    int K, int N) {
  size_t i = (size_t)blockIdx.x * 256 + threadIdx.x;
  if (i >= (size_t)K * N) return;
  int n = (int)(i / K);
  int k = (int)(i % K);
  Wt[i] = (_Float16)W[(size_t)k * N + n];
}

// ---------------- LayerNorm (one block per row of 768) ----------------
__global__ __launch_bounds__(256) void ln_kernel(const float* __restrict__ x,
                                                 const float* __restrict__ w,
                                                 const float* __restrict__ bb,
                                                 _Float16* __restrict__ out) {
  const int row = blockIdx.x;
  const int tid = threadIdx.x;
  const float* xr = x + (size_t)row * DIM;
  float v0 = xr[tid], v1 = xr[tid + 256], v2 = xr[tid + 512];
  float s  = v0 + v1 + v2;
  float s2 = v0 * v0 + v1 * v1 + v2 * v2;
  #pragma unroll
  for (int m = 1; m < 32; m <<= 1) {
    s  += __shfl_xor(s,  m, 32);
    s2 += __shfl_xor(s2, m, 32);
  }
  __shared__ float red[2][8];
  if ((tid & 31) == 0) { red[0][tid >> 5] = s; red[1][tid >> 5] = s2; }
  __syncthreads();
  s = 0.f; s2 = 0.f;
  #pragma unroll
  for (int i = 0; i < 8; i++) { s += red[0][i]; s2 += red[1][i]; }
  float mu  = s * (1.0f / DIM);
  float var = s2 * (1.0f / DIM) - mu * mu;
  float rs  = rsqrtf(var + EPSF);
  _Float16* orow = out + (size_t)row * DIM;
  orow[tid]       = (_Float16)((v0 - mu) * rs * w[tid]       + bb[tid]);
  orow[tid + 256] = (_Float16)((v1 - mu) * rs * w[tid + 256] + bb[tid + 256]);
  orow[tid + 512] = (_Float16)((v2 - mu) * rs * w[tid + 512] + bb[tid + 512]);
}

// ---------------- generic WMMA GEMM: C = A[M,K] * Wt[N,K]^T + bias (+gelu)(+res) ----------------
// 256 threads = 8 waves; wave tile 64x32 (4 M-tiles x 2 N-tiles); block tile 128x128.
__global__ __launch_bounds__(256) void gemm_kernel(const _Float16* __restrict__ A,
                                                   const _Float16* __restrict__ BT,
                                                   const float* __restrict__ bias,
                                                   const float* __restrict__ res,
                                                   float* __restrict__ outF,
                                                   _Float16* __restrict__ outH,
                                                   int M, int N, int K, int gelu) {
  const int lane = threadIdx.x & 31;
  const int wv   = threadIdx.x >> 5;
  const int wm   = wv >> 2, wn = wv & 3;
  const int hl   = lane >> 4, l15 = lane & 15;
  const int n0   = blockIdx.x * 128 + wn * 32;
  const int m0   = blockIdx.y * 128 + wm * 64;

  v8f acc[4][2] = {};

  const _Float16* arow[4];
  #pragma unroll
  for (int mt = 0; mt < 4; mt++) {
    int m = m0 + mt * 16 + l15;
    if (m > M - 1) m = M - 1;       // clamp tail rows (results discarded at store)
    arow[mt] = A + (size_t)m * K;
  }
  const _Float16* brow[2];
  #pragma unroll
  for (int nt = 0; nt < 2; nt++) {
    int n = n0 + nt * 16 + l15;     // N is always a multiple of 128 here
    brow[nt] = BT + (size_t)n * K;
  }

  for (int k0 = 0; k0 < K; k0 += 32) {
    v16h b0 = ld_btfrag(brow[0] + k0, hl);
    v16h b1 = ld_btfrag(brow[1] + k0, hl);
    #pragma unroll
    for (int mt = 0; mt < 4; mt++) {
      v16h af = ld_afrag(arow[mt] + k0, hl);
      acc[mt][0] = WMMA(af, b0, acc[mt][0]);
      acc[mt][1] = WMMA(af, b1, acc[mt][1]);
    }
  }

  #pragma unroll
  for (int mt = 0; mt < 4; mt++) {
    #pragma unroll
    for (int j = 0; j < 8; j++) {
      int mm = m0 + mt * 16 + j + 8 * hl;   // C/D layout: VGPR j -> M = j + 8*hl
      if (mm < M) {
        #pragma unroll
        for (int nt = 0; nt < 2; nt++) {
          int n = n0 + nt * 16 + l15;       // C/D layout: lane -> N = l&15
          float v = acc[mt][nt][j] + bias[n];
          if (gelu) v = 0.5f * v * (1.0f + erff(v * 0.70710678118f));
          size_t idx = (size_t)mm * N + n;
          if (res)  v += res[idx];
          if (outF) outF[idx] = v;
          if (outH) outH[idx] = (_Float16)v;
        }
      }
    }
  }
}

// ---------------- repack V: vt[(b*NH+h)*HD + d][key] f16, zero-padded keys ----------------
__global__ __launch_bounds__(256) void packv_kernel(const _Float16* __restrict__ qkv,
                                                    _Float16* __restrict__ vt) {
  int i = blockIdx.x * 256 + threadIdx.x;
  if (i >= BB * NH * HD * TPAD) return;
  int t  = i % TPAD;
  int r  = i / TPAD;
  int d  = r % HD;
  int bh = r / HD;
  int b  = bh / NH, h = bh % NH;
  _Float16 v = (_Float16)0.f;
  if (t < TT) v = qkv[(size_t)(b * TT + t) * (3 * DIM) + 2 * DIM + h * HD + d];
  vt[i] = v;
}

// ---------------- attention: one wave per (b, h, 16-query tile), flash-style ----------------
__global__ __launch_bounds__(128) void attn_kernel(const _Float16* __restrict__ qkv,
                                                   const _Float16* __restrict__ vt,
                                                   _Float16* __restrict__ o) {
  __shared__ __align__(16) _Float16 pbuf[4][16][32];   // per-wave P tile (C-layout -> A-layout bounce)
  const int lane = threadIdx.x & 31;
  const int wv   = threadIdx.x >> 5;
  const int w    = blockIdx.x * 4 + wv;
  const int bh   = w / QTILES;
  const int qt   = w % QTILES;
  const int b    = bh / NH, h = bh % NH;
  const int hl   = lane >> 4, l15 = lane & 15;

  // Q fragments (16 queries x 64 dims = two 16x32 A-frags); clamp padded rows
  int qrow = qt * 16 + l15;
  if (qrow > TT - 1) qrow = TT - 1;
  const _Float16* qp = qkv + (size_t)(b * TT + qrow) * (3 * DIM) + h * HD;
  v16h qa0 = ld_afrag(qp, hl);
  v16h qa1 = ld_afrag(qp + 32, hl);

  v8f acc0 = {}, acc1 = {}, acc2 = {}, acc3 = {};
  float mrun[8], lsum[8];
  #pragma unroll
  for (int j = 0; j < 8; j++) { mrun[j] = -3.0e38f; lsum[j] = 0.f; }

  _Float16* pb = &pbuf[wv][0][0];

  for (int kc = 0; kc < TPAD; kc += 32) {
    // ---- S = (Q K^T) * scale for a 32-key chunk (two 16-key N-tiles) ----
    v8f s0 = {}, s1 = {};
    int k0 = kc + l15;        int kr0 = (k0 > TT - 1) ? TT - 1 : k0;
    int k1 = kc + 16 + l15;   int kr1 = (k1 > TT - 1) ? TT - 1 : k1;
    const _Float16* kp0 = qkv + (size_t)(b * TT + kr0) * (3 * DIM) + DIM + h * HD;
    const _Float16* kp1 = qkv + (size_t)(b * TT + kr1) * (3 * DIM) + DIM + h * HD;
    s0 = WMMA(qa0, ld_btfrag(kp0, hl),      s0);
    s0 = WMMA(qa1, ld_btfrag(kp0 + 32, hl), s0);
    s1 = WMMA(qa0, ld_btfrag(kp1, hl),      s1);
    s1 = WMMA(qa1, ld_btfrag(kp1 + 32, hl), s1);

    const bool msk0 = (k0 >= TT), msk1 = (k1 >= TT);
    float mt[8];
    #pragma unroll
    for (int j = 0; j < 8; j++) {
      float a0 = s0[j] * 0.125f; if (msk0) a0 = -3.0e38f; s0[j] = a0;
      float a1 = s1[j] * 0.125f; if (msk1) a1 = -3.0e38f; s1[j] = a1;
      mt[j] = fmaxf(a0, a1);
    }
    // row max across the 16 lanes holding the same query row
    #pragma unroll
    for (int xm = 1; xm < 16; xm <<= 1) {
      #pragma unroll
      for (int j = 0; j < 8; j++) mt[j] = fmaxf(mt[j], __shfl_xor(mt[j], xm, 32));
    }
    // online-softmax update; write P tile to LDS in [row][key] layout
    #pragma unroll
    for (int j = 0; j < 8; j++) {
      float mnew = fmaxf(mrun[j], mt[j]);
      float corr = __expf(mrun[j] - mnew);
      float p0   = __expf(s0[j] - mnew);
      float p1   = __expf(s1[j] - mnew);
      lsum[j] = lsum[j] * corr + p0 + p1;
      mrun[j] = mnew;
      acc0[j] *= corr; acc1[j] *= corr; acc2[j] *= corr; acc3[j] *= corr;
      pb[(j + 8 * hl) * 32 + l15]      = (_Float16)p0;
      pb[(j + 8 * hl) * 32 + 16 + l15] = (_Float16)p1;
    }
    asm volatile("s_wait_dscnt 0" ::: "memory");   // order LDS store -> load within the wave
    v16h pf = ld_afrag(pb + l15 * 32, hl);

    // ---- O += P @ V  (V in dim-major vt, padded keys are zero) ----
    const _Float16* vp = vt + ((size_t)bh * HD + l15) * TPAD + kc;
    acc0 = WMMA(pf, ld_btfrag(vp,               hl), acc0);
    acc1 = WMMA(pf, ld_btfrag(vp + 16 * TPAD,   hl), acc1);
    acc2 = WMMA(pf, ld_btfrag(vp + 32 * TPAD,   hl), acc2);
    acc3 = WMMA(pf, ld_btfrag(vp + 48 * TPAD,   hl), acc3);
  }

  // reduce l across the 16-lane row group, normalize, store
  #pragma unroll
  for (int xm = 1; xm < 16; xm <<= 1) {
    #pragma unroll
    for (int j = 0; j < 8; j++) lsum[j] += __shfl_xor(lsum[j], xm, 32);
  }
  #pragma unroll
  for (int j = 0; j < 8; j++) {
    int q = qt * 16 + j + 8 * hl;
    if (q < TT) {
      float r = 1.0f / lsum[j];
      size_t base = (size_t)(b * TT + q) * DIM + h * HD + l15;
      o[base +  0] = (_Float16)(acc0[j] * r);
      o[base + 16] = (_Float16)(acc1[j] * r);
      o[base + 32] = (_Float16)(acc2[j] * r);
      o[base + 48] = (_Float16)(acc3[j] * r);
    }
  }
}

// ---------------- host orchestration ----------------
extern "C" void kernel_launch(void* const* d_in, const int* in_sizes, int n_in,
                              void* d_out, int out_size, void* d_ws, size_t ws_size,
                              hipStream_t stream) {
  (void)in_sizes; (void)n_in; (void)out_size; (void)ws_size;
  const float* x     = (const float*)d_in[0];
  const float* ln1w  = (const float*)d_in[1];
  const float* ln1b  = (const float*)d_in[2];
  const float* qkvw  = (const float*)d_in[3];
  const float* qkvb  = (const float*)d_in[4];
  const float* projw = (const float*)d_in[5];
  const float* projb = (const float*)d_in[6];
  const float* ln2w  = (const float*)d_in[7];
  const float* ln2b  = (const float*)d_in[8];
  const float* fc1w  = (const float*)d_in[9];
  const float* fc1b  = (const float*)d_in[10];
  const float* fc2w  = (const float*)d_in[11];
  const float* fc2b  = (const float*)d_in[12];

  char* ws = (char*)d_ws;
  auto carve = [&](size_t bytes) -> char* {
    char* p = ws;
    ws += (bytes + 255) & ~(size_t)255;
    return p;
  };
  _Float16* wt_qkv  = (_Float16*)carve((size_t)DIM * 3 * DIM * 2);
  _Float16* wt_proj = (_Float16*)carve((size_t)DIM * DIM * 2);
  _Float16* wt_fc1  = (_Float16*)carve((size_t)DIM * HIDDEN * 2);
  _Float16* wt_fc2  = (_Float16*)carve((size_t)HIDDEN * DIM * 2);
  _Float16* h16     = (_Float16*)carve((size_t)MROWS * DIM * 2);
  _Float16* qkv16   = (_Float16*)carve((size_t)MROWS * 3 * DIM * 2);
  _Float16* vt      = (_Float16*)carve((size_t)BB * NH * HD * TPAD * 2);
  _Float16* attn16  = (_Float16*)carve((size_t)MROWS * DIM * 2);
  float*    x1      = (float*)   carve((size_t)MROWS * DIM * 4);
  // reuse: GELU activations overlay qkv16(+vt) region (both dead by then); LN2 output overlays h16
  _Float16* gelu16  = qkv16;
  _Float16* h2      = h16;

  dim3 blk(256);
  const int mb = (MROWS + 127) / 128;

  // f16 transposed weights
  convT_kernel<<<dim3((DIM * 3 * DIM + 255) / 256), blk, 0, stream>>>(qkvw,  wt_qkv,  DIM,    3 * DIM);
  convT_kernel<<<dim3((DIM * DIM     + 255) / 256), blk, 0, stream>>>(projw, wt_proj, DIM,    DIM);
  convT_kernel<<<dim3((DIM * HIDDEN  + 255) / 256), blk, 0, stream>>>(fc1w,  wt_fc1,  DIM,    HIDDEN);
  convT_kernel<<<dim3((HIDDEN * DIM  + 255) / 256), blk, 0, stream>>>(fc2w,  wt_fc2,  HIDDEN, DIM);

  // h = LN1(x)  (f16)
  ln_kernel<<<dim3(MROWS), blk, 0, stream>>>(x, ln1w, ln1b, h16);
  // qkv = h @ qkv_w + qkv_b  (f16)
  gemm_kernel<<<dim3(3 * DIM / 128, mb), blk, 0, stream>>>(h16, wt_qkv, qkvb, nullptr, nullptr, qkv16,
                                                           MROWS, 3 * DIM, DIM, 0);
  // V -> dim-major, zero-padded keys
  packv_kernel<<<dim3((BB * NH * HD * TPAD + 255) / 256), blk, 0, stream>>>(qkv16, vt);
  // attention (one wave per (b,h,qtile); 16*12*37 waves / 4 per block)
  attn_kernel<<<dim3(BB * NH * QTILES / 4), dim3(128), 0, stream>>>(qkv16, vt, attn16);
  // x1 = x + attn @ proj_w + proj_b  (f32)
  gemm_kernel<<<dim3(DIM / 128, mb), blk, 0, stream>>>(attn16, wt_proj, projb, x, x1, nullptr,
                                                       MROWS, DIM, DIM, 0);
  // h2 = LN2(x1)  (f16)
  ln_kernel<<<dim3(MROWS), blk, 0, stream>>>(x1, ln2w, ln2b, h2);
  // g = gelu(h2 @ fc1_w + fc1_b)  (f16)
  gemm_kernel<<<dim3(HIDDEN / 128, mb), blk, 0, stream>>>(h2, wt_fc1, fc1b, nullptr, nullptr, gelu16,
                                                          MROWS, HIDDEN, DIM, 1);
  // out = x1 + g @ fc2_w + fc2_b  (f32)
  gemm_kernel<<<dim3(DIM / 128, mb), blk, 0, stream>>>(gelu16, wt_fc2, fc2b, x1, (float*)d_out, nullptr,
                                                       MROWS, DIM, HIDDEN, 0);
}
